// GRNDecoder_24567212933620
// MI455X (gfx1250) — matrix operand order repaired and checked by reading, not compiled, not run on hardware
//
#include <hip/hip_runtime.h>

// ---------------- problem constants ----------------
#define B_    4
#define N_    2000
#define FG    16
#define H_    512
#define T_    30
#define E_    64000
#define K3H   1536            // 3*H
#define KDIM  32000           // N*FG (GEMM reduction dim)
#define KS    25              // K-splits for big GEMM
#define KCH   (KDIM / KS)     // 1280, multiple of 64
#define NT_IH (K3H / 16)      // 96 N-tiles
#define NT_FC (N_ / 16)       // 125 N-tiles
#define IT_IH ((KCH - 64) / 64)   // 19 steady-state iterations
#define IT_FC ((H_ - 64) / 64)    // 7 steady-state iterations

typedef unsigned short u16;
typedef __attribute__((ext_vector_type(16))) __bf16 v16bf;
typedef __attribute__((ext_vector_type(8)))  float  v8f;

union V16U {
    v16bf v;
    uint4 q[2];
    u16   s[16];
};

__device__ __forceinline__ u16 f2bf(float f) {
    unsigned int u = __float_as_uint(f);
    unsigned int r = (u + 0x7FFFu + ((u >> 16) & 1u)) >> 16;   // RNE
    return (u16)r;
}

// A fragment (ISA 16-bit A 16x32 layout): p = row + half*8 (+k); K {p..p+7, p+16..p+23}
__device__ __forceinline__ void ldA(V16U& a, const u16* __restrict__ p) {
    a.q[0] = *(const uint4*)(p);
    a.q[1] = *(const uint4*)(p + 16);
}
// B fragment (bf16): p = row + half*8 (+k); K {p..p+7, p+16..p+23}
__device__ __forceinline__ void ldB(V16U& b, const u16* __restrict__ p) {
    b.q[0] = *(const uint4*)(p);
    b.q[1] = *(const uint4*)(p + 16);
}
// B fragment from f32 source, converted in-register (fallback path)
__device__ __forceinline__ void ldBf(V16U& b, const float* __restrict__ p) {
#pragma unroll
    for (int i = 0; i < 16; i += 4) {
        float4 f = *(const float4*)(p + i + (i >= 8 ? 8 : 0));
        b.s[i + 0] = f2bf(f.x);
        b.s[i + 1] = f2bf(f.y);
        b.s[i + 2] = f2bf(f.z);
        b.s[i + 3] = f2bf(f.w);
    }
}

#define WMMA_BF16(C, A, Bv) \
    __builtin_amdgcn_wmma_f32_16x16x32_bf16(false, (A), false, (Bv), (short)0, (C), false, false)

// ---------------- setup kernels ----------------
__global__ void k_zero_i32(int* p, int n) {
    int i = blockIdx.x * blockDim.x + threadIdx.x;
    if (i < n) p[i] = 0;
}
__global__ void k_zero_f32(float* p, int n) {
    int i = blockIdx.x * blockDim.x + threadIdx.x;
    if (i < n) p[i] = 0.0f;
}
__global__ void k_deg(const int* __restrict__ dst, int* __restrict__ deg) {
    int e = blockIdx.x * blockDim.x + threadIdx.x;
    if (e < E_) atomicAdd(&deg[dst[e]], 1);
}
__global__ void k_dinv(const int* __restrict__ deg, float* __restrict__ dinv) {
    int n = blockIdx.x * blockDim.x + threadIdx.x;
    if (n < N_) dinv[n] = rsqrtf((float)(deg[n] + 1));   // +1 = self loop
}
// f32 -> bf16 weight cache (4 elems/thread)
__global__ void k_cvt(const float* __restrict__ in, u16* __restrict__ out, long long n) {
    long long i = ((long long)blockIdx.x * blockDim.x + threadIdx.x) * 4;
    if (i + 3 < n) {
        float4 f = *(const float4*)(in + i);
        uint2 v;
        v.x = (unsigned int)f2bf(f.x) | ((unsigned int)f2bf(f.y) << 16);
        v.y = (unsigned int)f2bf(f.z) | ((unsigned int)f2bf(f.w) << 16);
        *(uint2*)(out + i) = v;
    } else {
        for (; i < n; ++i) out[i] = f2bf(in[i]);
    }
}

// ---------------- per-timestep kernels ----------------
// GCN message scatter: s[b,dst] += dinv[src]*dinv[dst]*x[b,src]
__global__ void k_scatter(const int* __restrict__ src, const int* __restrict__ dst,
                          const float* __restrict__ dinv, const float* __restrict__ x,
                          float* __restrict__ s) {
    int idx = blockIdx.x * blockDim.x + threadIdx.x;
    if (idx >= B_ * E_) return;
    int b = idx / E_, e = idx - b * E_;
    int sn = src[e], dn = dst[e];
    atomicAdd(&s[b * N_ + dn], dinv[sn] * dinv[dn] * x[b * N_ + sn]);
}

// emb[b,n,f] = relu((s + dinv^2*x)*W_gcn[f] + b_gcn[f]) -> bf16, k = n*16+f
// emb buffer is padded to 16 rows; rows 4..15 are zeroed once at setup.
__global__ void k_emb(const float* __restrict__ s, const float* __restrict__ dinv,
                      const float* __restrict__ x, const float* __restrict__ Wg,
                      const float* __restrict__ bg, u16* __restrict__ emb) {
    int idx = blockIdx.x * blockDim.x + threadIdx.x;
    if (idx >= B_ * N_) return;
    int b = idx / N_, n = idx - b * N_;
    float di    = dinv[n];
    float total = s[idx] + di * di * x[idx];
    V16U o;
#pragma unroll
    for (int f = 0; f < 16; ++f) {
        float v = fmaf(total, Wg[f], bg[f]);
        o.s[f] = f2bf(v > 0.0f ? v : 0.0f);
    }
    uint4* dp = (uint4*)(emb + (size_t)b * KDIM + n * 16);
    dp[0] = o.q[0];
    dp[1] = o.q[1];
}

// Big GEMM: gi partials = emb(16x32000, rows>=4 zero-padded) @ W_ih^T.
// bf16 WMMA, f32 acc, 2-stage software pipeline, pointer-bump addressing,
// compile-time trip count. One wave per (tile, ksplit).
template <bool CONV>
__global__ __launch_bounds__(256) void k_gemm_ih(const u16* __restrict__ emb_bf,
                                                 const u16* __restrict__ w_bf,
                                                 const float* __restrict__ w_f,
                                                 float* __restrict__ part) {
    const int wave = blockIdx.x * (blockDim.x >> 5) + (threadIdx.x >> 5);
    if (wave >= NT_IH * KS) return;                      // wave-uniform
    const int tile = wave % NT_IH;
    const int ks   = wave / NT_IH;
    const int lane = threadIdx.x & 31;
    const int m    = lane & 15;
    const int half = lane >> 4;
    const int kstart = ks * KCH;

    v8f c = {};
    const u16*   pa  = emb_bf + (size_t)m * KDIM + half * 8 + kstart;
    const size_t boff = (size_t)(tile * 16 + m) * KDIM;
    const u16*   pb  = w_bf + boff + half * 8 + kstart;
    const float* pf  = w_f + boff + half * 8 + kstart;

    V16U a0, b0, a1, b1;
    ldA(a0, pa);
    ldA(a1, pa + 32);
    if (CONV) { ldB(b0, pb); ldB(b1, pb + 32); }
    else      { ldBf(b0, pf); ldBf(b1, pf + 32); }

    for (int it = IT_IH; it > 0; --it) {
        c = WMMA_BF16(c, a0.v, b0.v);
        pa += 64; pb += 64; pf += 64;                    // independent VALU fills hazard slots
        ldA(a0, pa);
        if (CONV) { ldB(b0, pb); __builtin_prefetch(pb + 512, 0, 1); }
        else      { ldBf(b0, pf); __builtin_prefetch(pf + 512, 0, 1); }
        c = WMMA_BF16(c, a1.v, b1.v);
        ldA(a1, pa + 32);
        if (CONV) ldB(b1, pb + 32); else ldBf(b1, pf + 32);
    }
    c = WMMA_BF16(c, a0.v, b0.v);
    c = WMMA_BF16(c, a1.v, b1.v);

    if (half == 0) {                                     // D rows 0..3 = VGPR 0..3, lanes 0..15
#pragma unroll
        for (int r = 0; r < 4; ++r)
            part[(size_t)(ks * 4 + r) * K3H + tile * 16 + lane] = c[r];
    }
}

// GRU gates: reduce gi partials, small h@W_hh^T dots, update h (f32 + bf16 copy).
__global__ __launch_bounds__(256) void k_gate(const float* __restrict__ part,
                                              const float* __restrict__ bih,
                                              const float* __restrict__ Whh,
                                              const float* __restrict__ bhh,
                                              const float* __restrict__ hp,
                                              float* __restrict__ hnext,
                                              u16* __restrict__ hbf) {
    int idx = blockIdx.x * blockDim.x + threadIdx.x;     // B_*H_ = 2048 threads
    int b = idx >> 9, i = idx & 511;
    float gir = bih[i], giz = bih[i + H_], gin = bih[i + 2 * H_];
    for (int ks = 0; ks < KS; ++ks) {
        const float* p = part + (size_t)(ks * 4 + b) * K3H;
        gir += p[i];
        giz += p[i + H_];
        gin += p[i + 2 * H_];
    }
    float ghr = bhh[i], ghz = bhh[i + H_], ghn = bhh[i + 2 * H_];
    const float* hrow = hp + b * H_;
    const float* w0 = Whh + (size_t)i * H_;
    const float* w1 = Whh + (size_t)(i + H_) * H_;
    const float* w2 = Whh + (size_t)(i + 2 * H_) * H_;
    for (int k = 0; k < H_; ++k) {
        float hk = hrow[k];
        ghr = fmaf(hk, w0[k], ghr);
        ghz = fmaf(hk, w1[k], ghz);
        ghn = fmaf(hk, w2[k], ghn);
    }
    float r  = 1.0f / (1.0f + __expf(-(gir + ghr)));
    float z  = 1.0f / (1.0f + __expf(-(giz + ghz)));
    float nn = tanhf(gin + r * ghn);
    float hv = (1.0f - z) * nn + z * hrow[i];
    hnext[idx] = hv;
    hbf[idx]   = f2bf(hv);
}

// FC: out = h @ W_fc^T + b_fc via bf16 WMMA (h padded to 16 rows, 4..15 zero).
template <bool CONV>
__global__ __launch_bounds__(256) void k_fc(const u16* __restrict__ h_bf,
                                            const u16* __restrict__ wfc_bf,
                                            const float* __restrict__ wfc_f,
                                            const float* __restrict__ bfc,
                                            float* __restrict__ out,
                                            float* __restrict__ xnext, int t) {
    const int wave = blockIdx.x * (blockDim.x >> 5) + (threadIdx.x >> 5);
    if (wave >= NT_FC) return;                           // wave-uniform
    const int lane = threadIdx.x & 31;
    const int m    = lane & 15;
    const int half = lane >> 4;

    v8f c = {};
    const u16*   pa  = h_bf + (size_t)m * H_ + half * 8;
    const size_t boff = (size_t)(wave * 16 + m) * H_;
    const u16*   pb  = wfc_bf + boff + half * 8;
    const float* pf  = wfc_f + boff + half * 8;

    V16U a0, b0, a1, b1;
    ldA(a0, pa);
    ldA(a1, pa + 32);
    if (CONV) { ldB(b0, pb); ldB(b1, pb + 32); }
    else      { ldBf(b0, pf); ldBf(b1, pf + 32); }

    for (int it = IT_FC; it > 0; --it) {
        c = WMMA_BF16(c, a0.v, b0.v);
        pa += 64; pb += 64; pf += 64;
        ldA(a0, pa);
        if (CONV) ldB(b0, pb); else ldBf(b0, pf);
        c = WMMA_BF16(c, a1.v, b1.v);
        ldA(a1, pa + 32);
        if (CONV) ldB(b1, pb + 32); else ldBf(b1, pf + 32);
    }
    c = WMMA_BF16(c, a0.v, b0.v);
    c = WMMA_BF16(c, a1.v, b1.v);

    if (half == 0) {
        int n = wave * 16 + lane;
        float bias = bfc[n];
#pragma unroll
        for (int r = 0; r < 4; ++r) {
            float v = c[r] + bias;
            out[(size_t)r * (T_ * N_) + (size_t)t * N_ + n] = v;  // (B,T,N)
            xnext[r * N_ + n] = v;
        }
    }
}

// ---------------- host launcher ----------------
extern "C" void kernel_launch(void* const* d_in, const int* in_sizes, int n_in,
                              void* d_out, int out_size, void* d_ws, size_t ws_size,
                              hipStream_t stream) {
    const float* x_in  = (const float*)d_in[0];
    const float* h_in  = (const float*)d_in[1];
    const int*   ei    = (const int*)d_in[2];
    const float* W_gcn = (const float*)d_in[3];
    const float* b_gcn = (const float*)d_in[4];
    const float* W_ih  = (const float*)d_in[5];
    const float* b_ih  = (const float*)d_in[6];
    const float* W_hh  = (const float*)d_in[7];
    const float* b_hh  = (const float*)d_in[8];
    const float* W_fc  = (const float*)d_in[9];
    const float* b_fc  = (const float*)d_in[10];
    float*       out   = (float*)d_out;

    char*  ws  = (char*)d_ws;
    size_t off = 0;
    auto alloc = [&](size_t bytes) -> char* {
        char* p = ws + off;
        off = (off + bytes + 255) & ~(size_t)255;
        return p;
    };
    float* s_buf  = (float*)alloc((size_t)B_ * N_ * 4);
    float* dinv   = (float*)alloc((size_t)N_ * 4);
    int*   deg    = (int*)alloc((size_t)N_ * 4);
    float* x0     = (float*)alloc((size_t)B_ * N_ * 4);
    float* x1     = (float*)alloc((size_t)B_ * N_ * 4);
    float* h0     = (float*)alloc((size_t)B_ * H_ * 4);
    float* h1     = (float*)alloc((size_t)B_ * H_ * 4);
    u16*   h_bf   = (u16*)alloc((size_t)16 * H_ * 2);        // padded to 16 rows
    u16*   emb_bf = (u16*)alloc((size_t)16 * KDIM * 2);      // padded to 16 rows
    float* part   = (float*)alloc((size_t)KS * 4 * K3H * 4);
    u16*   wfc_bf = (u16*)alloc((size_t)N_ * H_ * 2);
    u16*   wih_bf = (u16*)alloc((size_t)K3H * KDIM * 2);
    (void)alloc(512);                                        // tail guard
    const bool conv = (ws_size >= off);   // ~102.1 MB: bf16 weight cache fits?

    // ---- one-time (per launch) setup ----
    hipMemcpyAsync(x0, x_in, (size_t)B_ * N_ * 4, hipMemcpyDeviceToDevice, stream);
    hipMemcpyAsync(h0, h_in, (size_t)B_ * H_ * 4, hipMemcpyDeviceToDevice, stream);
    k_zero_i32<<<(N_ + 255) / 256, 256, 0, stream>>>(deg, N_);
    // zero padded rows 4..15 of emb/h bf16 A-buffers (stay zero across steps)
    k_zero_i32<<<(12 * KDIM / 2 + 255) / 256, 256, 0, stream>>>(
        (int*)(emb_bf + (size_t)4 * KDIM), 12 * KDIM / 2);
    k_zero_i32<<<(12 * H_ / 2 + 255) / 256, 256, 0, stream>>>(
        (int*)(h_bf + (size_t)4 * H_), 12 * H_ / 2);
    k_deg<<<(E_ + 255) / 256, 256, 0, stream>>>(ei + E_, deg);
    k_dinv<<<(N_ + 255) / 256, 256, 0, stream>>>(deg, dinv);
    if (conv) {
        long long n_ih = (long long)K3H * KDIM;
        long long n_fc = (long long)N_ * H_;
        k_cvt<<<(unsigned)((n_ih / 4 + 255) / 256), 256, 0, stream>>>(W_ih, wih_bf, n_ih);
        k_cvt<<<(unsigned)((n_fc / 4 + 255) / 256), 256, 0, stream>>>(W_fc, wfc_bf, n_fc);
    }

    float *xc = x0, *xn = x1, *hc = h0, *hn = h1;
    for (int t = 0; t < T_; ++t) {
        k_zero_f32<<<(B_ * N_ + 255) / 256, 256, 0, stream>>>(s_buf, B_ * N_);
        k_scatter<<<(B_ * E_ + 255) / 256, 256, 0, stream>>>(ei, ei + E_, dinv, xc, s_buf);
        k_emb<<<(B_ * N_ + 255) / 256, 256, 0, stream>>>(s_buf, dinv, xc, W_gcn, b_gcn, emb_bf);
        if (conv)
            k_gemm_ih<true><<<(NT_IH * KS) / 8, 256, 0, stream>>>(emb_bf, wih_bf, W_ih, part);
        else
            k_gemm_ih<false><<<(NT_IH * KS) / 8, 256, 0, stream>>>(emb_bf, wih_bf, W_ih, part);
        k_gate<<<(B_ * H_) / 256, 256, 0, stream>>>(part, b_ih, W_hh, b_hh, hc, hn, h_bf);
        if (conv)
            k_fc<true><<<(NT_FC + 7) / 8, 256, 0, stream>>>(h_bf, wfc_bf, W_fc, b_fc, out, xn, t);
        else
            k_fc<false><<<(NT_FC + 7) / 8, 256, 0, stream>>>(h_bf, wfc_bf, W_fc, b_fc, out, xn, t);
        float* tf = xc; xc = xn; xn = tf;
        tf = hc; hc = hn; hn = tf;
    }
}
